// VRNN_74285754352062
// MI455X (gfx1250) — matrix operand order repaired and checked by reading, not compiled, not run on hardware
//
#include <hip/hip_runtime.h>

typedef __attribute__((ext_vector_type(16))) _Float16 v16h;
typedef __attribute__((ext_vector_type(8)))  _Float16 v8h;
typedef __attribute__((ext_vector_type(8)))  float    v8f;
typedef __attribute__((ext_vector_type(4)))  float    v4f;

#define SEQ   256
#define BATCH 128
#define XDIM  128
#define HDIM  1024
#define ZDIM  64

// ---------------------------------------------------------------------------
// Pack a row-major f32 weight W[N][K] into f16 WMMA B-fragment order:
// P[((nb*kTiles + kb)*32 + lane)*16 + i] = W[nb*16 + (lane&15)][kb*32 + (i>>3)*16 + (lane>>4)*8 + (i&7)]
// so each lane's 16-half B fragment for one 32xK-tile is one contiguous 32B load.
// ---------------------------------------------------------------------------
__global__ void pack_w_kernel(const float* __restrict__ W, _Float16* __restrict__ P,
                              int N, int K) {
  size_t idx = (size_t)blockIdx.x * blockDim.x + threadIdx.x;
  size_t tot = (size_t)N * K;
  if (idx >= tot) return;
  int    i    = (int)(idx & 15);
  int    lane = (int)((idx >> 4) & 31);
  size_t tile = idx >> 9;
  int kTiles  = K >> 5;
  int kb = (int)(tile % (size_t)kTiles);
  int nb = (int)(tile / (size_t)kTiles);
  int n = nb * 16 + (lane & 15);
  int k = kb * 32 + (i >> 3) * 16 + (lane >> 4) * 8 + (i & 7);
  P[idx] = (_Float16)W[(size_t)n * K + k];
}

// ---------------------------------------------------------------------------
// Fused GEMM: C = act(A @ W^T + bias),  A is BATCHxK (f16 or f32, row-major,
// leading dim ldA), W pre-packed f16 fragments.  One wave computes one 16x16
// C tile via v_wmma_f32_16x16x32_f16 over K.  4 waves / block cover 64 cols.
// act: 0 = none, 1 = relu, 2 = exp(0.5*x)
// Epilogue writes optional f32 dest + up to two f16 dests (free concat).
// ---------------------------------------------------------------------------
template<bool AF32>
__global__ void gemm_wmma_kernel(const void* __restrict__ Aptr, int ldA,
                                 const _Float16* __restrict__ Wp,
                                 const float* __restrict__ bias,
                                 int N, int K,
                                 float* __restrict__ Cf, int ldCf,
                                 _Float16* __restrict__ Ch0, int ldCh0,
                                 _Float16* __restrict__ Ch1, int ldCh1,
                                 int act) {
  const int lane  = threadIdx.x & 31;
  const int wave  = threadIdx.x >> 5;
  const int mb    = blockIdx.x;
  const int nb    = blockIdx.y * 4 + wave;
  const int grp   = lane >> 4;              // 0: K 0-7 & 16-23, 1: K 8-15 & 24-31
  const int mrow  = mb * 16 + (lane & 15);
  const int kTiles = K >> 5;
  const _Float16* Wbase = Wp + ((size_t)nb * kTiles * 32 + lane) * 16;

  v8f c = {};
  for (int kb = 0; kb < kTiles; ++kb) {
    v16h a;
    if (AF32) {
      const float* arow = (const float*)Aptr + (size_t)mrow * ldA + kb * 32 + grp * 8;
      v4f a0 = *(const v4f*)(arow);
      v4f a1 = *(const v4f*)(arow + 4);
      v4f a2 = *(const v4f*)(arow + 16);
      v4f a3 = *(const v4f*)(arow + 20);
      for (int j = 0; j < 4; ++j) {
        a[j]      = (_Float16)a0[j];
        a[4 + j]  = (_Float16)a1[j];
        a[8 + j]  = (_Float16)a2[j];
        a[12 + j] = (_Float16)a3[j];
      }
    } else {
      const _Float16* arow = (const _Float16*)Aptr + (size_t)mrow * ldA + kb * 32 + grp * 8;
      v8h lo = *(const v8h*)(arow);        // K = kb*32 + grp*8 .. +7
      v8h hi = *(const v8h*)(arow + 16);   // K = kb*32 + 16 + grp*8 .. +7
      for (int j = 0; j < 8; ++j) { a[j] = lo[j]; a[8 + j] = hi[j]; }
    }
    v16h b = *(const v16h*)(Wbase + (size_t)kb * 512);
    c = __builtin_amdgcn_wmma_f32_16x16x32_f16(false, a, false, b, (short)0, c,
                                               false, false);
  }

  const int ncol = nb * 16 + (lane & 15);
  const float bv = bias ? bias[ncol] : 0.0f;
#pragma unroll
  for (int r = 0; r < 8; ++r) {
    int m = mb * 16 + grp * 8 + r;         // C layout: lanes 0-15 -> M=r, 16-31 -> M=8+r
    float v = c[r] + bv;
    if (act == 1)      v = v > 0.0f ? v : 0.0f;
    else if (act == 2) v = __expf(0.5f * v);
    if (Cf) Cf[(size_t)m * ldCf + ncol] = v;
    if (Ch0 || Ch1) {
      _Float16 hv = (_Float16)v;
      if (Ch0) Ch0[(size_t)m * ldCh0 + ncol] = hv;
      if (Ch1) Ch1[(size_t)m * ldCh1 + ncol] = hv;
    }
  }
}

// ---------------------------------------------------------------------------
// elementwise kernels
// ---------------------------------------------------------------------------
__device__ __forceinline__ void block_reduce_add(float v, float* target) {
  __shared__ float red[256];
  int t = threadIdx.x;
  red[t] = v;
  __syncthreads();
  for (int s = 128; s > 0; s >>= 1) {
    if (t < s) red[t] += red[t + s];
    __syncthreads();
  }
  if (t == 0) atomicAdd(target, red[0]);
}

__global__ void z_kld_kernel(const float* __restrict__ enc_mean,
                             const float* __restrict__ enc_std,
                             const float* __restrict__ prior_mean,
                             const float* __restrict__ prior_std,
                             const float* __restrict__ eps_t,
                             _Float16* __restrict__ z16,
                             float* __restrict__ kld_acc) {
  int idx = blockIdx.x * blockDim.x + threadIdx.x;   // BATCH*ZDIM
  float em = enc_mean[idx], es = enc_std[idx];
  float pm = prior_mean[idx], ps = prior_std[idx];
  float z = em + eps_t[idx] * es;
  z16[idx] = (_Float16)z;
  float dm = em - pm;
  float rs = es / ps;
  float kld = -0.5f * (1.0f + 2.0f * __logf(es) - dm * dm - rs * rs);
  block_reduce_add(kld, kld_acc);
}

__global__ void gru_kernel(const float* __restrict__ gi, const float* __restrict__ gh,
                           float* __restrict__ h,
                           _Float16* __restrict__ h16a, _Float16* __restrict__ h16b) {
  int idx = blockIdx.x * blockDim.x + threadIdx.x;   // BATCH*HDIM
  int m = idx >> 10, n = idx & (HDIM - 1);
  const float* gir = gi + (size_t)m * (3 * HDIM);
  const float* ghr = gh + (size_t)m * (3 * HDIM);
  float r  = 1.0f / (1.0f + __expf(-(gir[n] + ghr[n])));
  float zg = 1.0f / (1.0f + __expf(-(gir[HDIM + n] + ghr[HDIM + n])));
  float nn = tanhf(gir[2 * HDIM + n] + r * ghr[2 * HDIM + n]);
  float hold = h[idx];
  float hnew = (1.0f - zg) * nn + zg * hold;
  h[idx] = hnew;
  _Float16 hv = (_Float16)hnew;
  h16a[(size_t)m * (2 * HDIM) + n] = hv;   // encin[:, HDIM:]
  h16b[(size_t)m * (2 * HDIM) + n] = hv;   // decin[:, HDIM:]
}

__global__ void recon_kernel(const float* __restrict__ logits,
                             const float* __restrict__ x_t,
                             float* __restrict__ xrec,
                             float* __restrict__ recon_acc) {
  int idx = blockIdx.x * blockDim.x + threadIdx.x;   // BATCH*XDIM
  float l = logits[idx], xv = x_t[idx];
  float sp_abs = log1pf(__expf(-fabsf(l)));
  float sp_pos = fmaxf(l, 0.0f) + sp_abs;            // softplus(l)
  float sp_neg = fmaxf(-l, 0.0f) + sp_abs;           // softplus(-l)
  float term = xv * sp_neg + (1.0f - xv) * sp_pos;
  xrec[idx] = 1.0f / (1.0f + __expf(-l));
  block_reduce_add(term, recon_acc);
}

__global__ void init_kernel(float* __restrict__ h,
                            _Float16* __restrict__ enc_h, _Float16* __restrict__ dec_h,
                            float* __restrict__ acc) {
  int idx = blockIdx.x * blockDim.x + threadIdx.x;
  if (idx < BATCH * HDIM) {
    h[idx] = 0.0f;
    int m = idx >> 10, n = idx & (HDIM - 1);
    enc_h[(size_t)m * (2 * HDIM) + n] = (_Float16)0.0f;
    dec_h[(size_t)m * (2 * HDIM) + n] = (_Float16)0.0f;
  }
  if (idx < 2) acc[idx] = 0.0f;
}

__global__ void finalize_kernel(const float* __restrict__ acc, float* __restrict__ out) {
  if (threadIdx.x == 0) {
    out[0] = acc[0] / (float)BATCH;
    out[1] = acc[1] / (float)BATCH;
  }
}

// ---------------------------------------------------------------------------
extern "C" void kernel_launch(void* const* d_in, const int* in_sizes, int n_in,
                              void* d_out, int out_size, void* d_ws, size_t ws_size,
                              hipStream_t stream) {
  const float* x            = (const float*)d_in[0];
  const float* eps          = (const float*)d_in[1];
  const float* phi_x_W      = (const float*)d_in[2];
  const float* phi_x_b      = (const float*)d_in[3];
  const float* enc_W        = (const float*)d_in[4];
  const float* enc_b        = (const float*)d_in[5];
  const float* enc_mean_W   = (const float*)d_in[6];
  const float* enc_mean_b   = (const float*)d_in[7];
  const float* enc_std_W    = (const float*)d_in[8];
  const float* enc_std_b    = (const float*)d_in[9];
  const float* prior_W      = (const float*)d_in[10];
  const float* prior_b      = (const float*)d_in[11];
  const float* prior_mean_W = (const float*)d_in[12];
  const float* prior_mean_b = (const float*)d_in[13];
  const float* prior_std_W  = (const float*)d_in[14];
  const float* prior_std_b  = (const float*)d_in[15];
  const float* phi_z_W      = (const float*)d_in[16];
  const float* phi_z_b      = (const float*)d_in[17];
  const float* dec1_W       = (const float*)d_in[18];
  const float* dec1_b       = (const float*)d_in[19];
  const float* dec2_W       = (const float*)d_in[20];
  const float* dec2_b       = (const float*)d_in[21];
  const float* gru_Wi       = (const float*)d_in[22];
  const float* gru_Wh       = (const float*)d_in[23];
  const float* gru_bi       = (const float*)d_in[24];
  const float* gru_bh       = (const float*)d_in[25];
  (void)in_sizes; (void)n_in; (void)out_size; (void)ws_size;

  float* out = (float*)d_out;

  char* ws = (char*)d_ws;
  size_t off = 0;
  auto alloc  = [&](size_t bytes) -> char* {
    char* p = ws + off;
    off = (off + bytes + 255) & ~((size_t)255);
    return p;
  };
  auto allocH = [&](size_t n) -> _Float16* { return (_Float16*)alloc(n * 2); };
  auto allocF = [&](size_t n) -> float*    { return (float*)alloc(n * 4); };

  // packed f16 weights (~30 MB total: all resident in 192 MB L2)
  _Float16* p_phi_x = allocH((size_t)HDIM * XDIM);
  _Float16* p_enc   = allocH((size_t)HDIM * 2 * HDIM);
  _Float16* p_enc_m = allocH((size_t)ZDIM * HDIM);
  _Float16* p_enc_s = allocH((size_t)ZDIM * HDIM);
  _Float16* p_prior = allocH((size_t)HDIM * HDIM);
  _Float16* p_pri_m = allocH((size_t)ZDIM * HDIM);
  _Float16* p_pri_s = allocH((size_t)ZDIM * HDIM);
  _Float16* p_phi_z = allocH((size_t)HDIM * ZDIM);
  _Float16* p_dec1  = allocH((size_t)HDIM * 2 * HDIM);
  _Float16* p_dec2  = allocH((size_t)XDIM * HDIM);
  _Float16* p_gWi   = allocH((size_t)3 * HDIM * 2 * HDIM);
  _Float16* p_gWh   = allocH((size_t)3 * HDIM * HDIM);

  // f16 activation buffers (interleaved for free concat)
  _Float16* encin   = allocH((size_t)BATCH * 2 * HDIM);  // [phi_x | h]
  _Float16* decin   = allocH((size_t)BATCH * 2 * HDIM);  // [phi_z | h]
  _Float16* ubuf    = allocH((size_t)BATCH * 2 * HDIM);  // [phi_x | phi_z]
  _Float16* prior16 = allocH((size_t)BATCH * HDIM);
  _Float16* enc16   = allocH((size_t)BATCH * HDIM);
  _Float16* dec16   = allocH((size_t)BATCH * HDIM);
  _Float16* z16     = allocH((size_t)BATCH * ZDIM);

  // f32 buffers
  float* f_pri_m  = allocF((size_t)BATCH * ZDIM);
  float* f_pri_s  = allocF((size_t)BATCH * ZDIM);
  float* f_enc_m  = allocF((size_t)BATCH * ZDIM);
  float* f_enc_s  = allocF((size_t)BATCH * ZDIM);
  float* f_gi     = allocF((size_t)BATCH * 3 * HDIM);
  float* f_gh     = allocF((size_t)BATCH * 3 * HDIM);
  float* f_logits = allocF((size_t)BATCH * XDIM);
  float* f_h      = allocF((size_t)BATCH * HDIM);
  float* f_acc    = allocF(2);   // [0]=recon, [1]=kld

  auto pack = [&](const float* W, _Float16* P, int N, int K) {
    size_t tot = (size_t)N * K;
    pack_w_kernel<<<dim3((unsigned)((tot + 255) / 256)), 256, 0, stream>>>(W, P, N, K);
  };
  pack(phi_x_W,      p_phi_x, HDIM, XDIM);
  pack(enc_W,        p_enc,   HDIM, 2 * HDIM);
  pack(enc_mean_W,   p_enc_m, ZDIM, HDIM);
  pack(enc_std_W,    p_enc_s, ZDIM, HDIM);
  pack(prior_W,      p_prior, HDIM, HDIM);
  pack(prior_mean_W, p_pri_m, ZDIM, HDIM);
  pack(prior_std_W,  p_pri_s, ZDIM, HDIM);
  pack(phi_z_W,      p_phi_z, HDIM, ZDIM);
  pack(dec1_W,       p_dec1,  HDIM, 2 * HDIM);
  pack(dec2_W,       p_dec2,  XDIM, HDIM);
  pack(gru_Wi,       p_gWi,   3 * HDIM, 2 * HDIM);
  pack(gru_Wh,       p_gWh,   3 * HDIM, HDIM);

  init_kernel<<<dim3(BATCH * HDIM / 256), 256, 0, stream>>>(f_h, encin + HDIM,
                                                            decin + HDIM, f_acc);

  auto gemmH = [&](const _Float16* A, int ldA, const _Float16* Wp, const float* bias,
                   int N, int K, int act,
                   float* Cf, int ldCf,
                   _Float16* Ch0, int ldCh0, _Float16* Ch1, int ldCh1) {
    dim3 grid(BATCH / 16, N / 64);
    gemm_wmma_kernel<false><<<grid, 128, 0, stream>>>(A, ldA, Wp, bias, N, K,
                                                      Cf, ldCf, Ch0, ldCh0, Ch1, ldCh1, act);
  };

  for (int t = 0; t < SEQ; ++t) {
    const float* x_t   = x   + (size_t)t * BATCH * XDIM;
    const float* eps_t = eps + (size_t)t * BATCH * ZDIM;

    // phi_x = relu(x_t @ phi_x_W^T + b) -> encin[:,0:H], u[:,0:H]   (f32 A, converted in-kernel)
    {
      dim3 grid(BATCH / 16, HDIM / 64);
      gemm_wmma_kernel<true><<<grid, 128, 0, stream>>>(x_t, XDIM, p_phi_x, phi_x_b,
                                                       HDIM, XDIM,
                                                       nullptr, 0, encin, 2 * HDIM,
                                                       ubuf, 2 * HDIM, 1);
    }
    // prior = relu(h @ prior_W^T + b)
    gemmH(encin + HDIM, 2 * HDIM, p_prior, prior_b, HDIM, HDIM, 1,
          nullptr, 0, prior16, HDIM, nullptr, 0);
    // enc = relu([phi_x|h] @ enc_W^T + b)
    gemmH(encin, 2 * HDIM, p_enc, enc_b, HDIM, 2 * HDIM, 1,
          nullptr, 0, enc16, HDIM, nullptr, 0);
    // prior_mean / prior_std = exp(0.5*lin)
    gemmH(prior16, HDIM, p_pri_m, prior_mean_b, ZDIM, HDIM, 0,
          f_pri_m, ZDIM, nullptr, 0, nullptr, 0);
    gemmH(prior16, HDIM, p_pri_s, prior_std_b, ZDIM, HDIM, 2,
          f_pri_s, ZDIM, nullptr, 0, nullptr, 0);
    // enc_mean / enc_std
    gemmH(enc16, HDIM, p_enc_m, enc_mean_b, ZDIM, HDIM, 0,
          f_enc_m, ZDIM, nullptr, 0, nullptr, 0);
    gemmH(enc16, HDIM, p_enc_s, enc_std_b, ZDIM, HDIM, 2,
          f_enc_s, ZDIM, nullptr, 0, nullptr, 0);
    // z = enc_mean + eps*enc_std ; accumulate KLD
    z_kld_kernel<<<dim3(BATCH * ZDIM / 256), 256, 0, stream>>>(f_enc_m, f_enc_s,
                                                               f_pri_m, f_pri_s,
                                                               eps_t, z16, f_acc + 1);
    // phi_z = relu(z @ phi_z_W^T + b) -> decin[:,0:H], u[:,H:2H]
    gemmH(z16, ZDIM, p_phi_z, phi_z_b, HDIM, ZDIM, 1,
          nullptr, 0, decin, 2 * HDIM, ubuf + HDIM, 2 * HDIM);
    // dec_h = relu([phi_z|h] @ dec1_W^T + b)
    gemmH(decin, 2 * HDIM, p_dec1, dec1_b, HDIM, 2 * HDIM, 1,
          nullptr, 0, dec16, HDIM, nullptr, 0);
    // logits = dec_h @ dec2_W^T + b
    gemmH(dec16, HDIM, p_dec2, dec2_b, XDIM, HDIM, 0,
          f_logits, XDIM, nullptr, 0, nullptr, 0);
    // gi = u @ gru_Wi^T + bi ; gh = h @ gru_Wh^T + bh  (biases fused in epilogue)
    gemmH(ubuf, 2 * HDIM, p_gWi, gru_bi, 3 * HDIM, 2 * HDIM, 0,
          f_gi, 3 * HDIM, nullptr, 0, nullptr, 0);
    gemmH(encin + HDIM, 2 * HDIM, p_gWh, gru_bh, 3 * HDIM, HDIM, 0,
          f_gh, 3 * HDIM, nullptr, 0, nullptr, 0);
    // GRU update -> h, encin[:,H:], decin[:,H:]
    gru_kernel<<<dim3(BATCH * HDIM / 256), 256, 0, stream>>>(f_gi, f_gh, f_h,
                                                             encin + HDIM, decin + HDIM);
    // recon loss + sigmoid(logits) -> x_recon slice
    recon_kernel<<<dim3(BATCH * XDIM / 256), 256, 0, stream>>>(
        f_logits, x_t, out + 2 + (size_t)t * BATCH * XDIM, f_acc);
  }

  finalize_kernel<<<1, 32, 0, stream>>>(f_acc, out);
}
